// ModalityCNNStem_14637248545453
// MI455X (gfx1250) — matrix operand order repaired and checked by reading, not compiled, not run on hardware
//
#include <hip/hip_runtime.h>
#include <hip/hip_bf16.h>
#include <math.h>

typedef _Float16 v4h  __attribute__((ext_vector_type(4)));
typedef _Float16 v8h  __attribute__((ext_vector_type(8)));
typedef _Float16 v16h __attribute__((ext_vector_type(16)));
typedef float    v8f  __attribute__((ext_vector_type(8)));

// Problem constants (from reference setup_inputs)
#define HCH 64      // hidden channels
#define PL  256     // patch length
#define DM  768     // d_model
#define KD  192     // im2col K dimension = HCH*3
#define STR 200     // padded LDS row stride (halfs); 400B rows -> 16B aligned, bank-spread

__device__ __forceinline__ float gelu_erf(float v) {
    // exact (erf) GELU, matching torch nn.GELU / jax approximate=False
    return 0.5f * v * (1.0f + erff(v * 0.70710678118654752f));
}

__global__ __launch_bounds__(256, 2)
void modality_stem_kernel(const float* __restrict__ patches,
                          const int*   __restrict__ types,
                          const float* __restrict__ W1, const float* __restrict__ b1,
                          const float* __restrict__ W2, const float* __restrict__ b2,
                          const float* __restrict__ W3, const float* __restrict__ b3,
                          float* __restrict__ out)
{
    extern __shared__ __align__(16) char smem[];
    _Float16* Bm     = (_Float16*)smem;          // [PL][STR]  transposed im2col activations (f16)
    _Float16* Ws     = Bm + PL * STR;            // [HCH][STR] W2 rows (f16)
    float*    xs     = (float*)(Ws + HCH * STR); // [PL+2] padded input signal
    float*    pooled = xs + 260;                 // [HCH] channel sums (16B aligned)
    float*    b2s    = pooled + HCH;             // [HCH] conv2 bias

    const int tid  = threadIdx.x;
    const int lane = tid & 31;
    const int wave = tid >> 5;
    const int m    = blockIdx.x;
    const int t    = types[m];

    // ---------------- Phase A: stage x, W2(f32->f16), biases into LDS ----------------
    if (tid < PL) xs[tid + 1] = patches[(size_t)m * PL + tid];
    if (tid == 0) { xs[0] = 0.f; xs[PL + 1] = 0.f; }
    if (tid < HCH) { pooled[tid] = 0.f; b2s[tid] = b2[t * HCH + tid]; }
    {
        const float4* w2v = (const float4*)(W2 + (size_t)t * HCH * KD);
        #pragma unroll
        for (int it = 0; it < 12; ++it) {
            const int f4 = it * 256 + tid;          // 3072 float4 total
            const float4 w = w2v[f4];
            const int f = f4 * 4;
            const int c = f / KD;
            const int k = f - c * KD;               // k%4==0 -> 8B-aligned b64 store
            v4h hv = { (_Float16)w.x, (_Float16)w.y, (_Float16)w.z, (_Float16)w.w };
            *(v4h*)&Ws[c * STR + k] = hv;
        }
    }
    __syncthreads();

    // ---------------- Phase B: stage1 conv(1->64,K=3)+GELU, scatter as im2col ----------------
    {
        const int ci = tid & 63;
        const int g  = tid >> 6;                    // 0..3 position groups of 64
        const float w0 = W1[(t * HCH + ci) * 3 + 0];
        const float w1 = W1[(t * HCH + ci) * 3 + 1];
        const float w2 = W1[(t * HCH + ci) * 3 + 2];
        const float bb = b1[t * HCH + ci];
        const int l0 = g * 64;
        float xm = xs[l0], x0 = xs[l0 + 1];
        #pragma unroll 4
        for (int j = 0; j < 64; ++j) {
            const int l = l0 + j;
            const float xp = xs[l + 2];
            const float v = fmaf(w0, xm, fmaf(w1, x0, fmaf(w2, xp, bb)));
            const _Float16 hv = (_Float16)gelu_erf(v);
            // Bm[row][ci*3+k] = h1[ci][row+k-1]  => h1[ci][l] lands at rows l+1, l, l-1
            if (l + 1 < PL) Bm[(l + 1) * STR + 3 * ci    ] = hv;
                            Bm[ l      * STR + 3 * ci + 1] = hv;
            if (l >= 1)     Bm[(l - 1) * STR + 3 * ci + 2] = hv;
            xm = x0; x0 = xp;
        }
        if (tid < HCH) {  // SAME-padding zeros at the two edges
            Bm[0 * STR + 3 * tid] = (_Float16)0.f;
            Bm[255 * STR + 3 * tid + 2] = (_Float16)0.f;
        }
    }
    // warm L2/L0 for the stage-3 W3 rows this thread will read
    __builtin_prefetch(W3 + ((size_t)t * DM + tid) * HCH, 0, 3);
    __syncthreads();

    // ---------------- Phase C: conv2 as WMMA GEMM  C[pos, ch] = A[pos,K] x B[K,ch] ----------------
    const int ct = wave & 3;          // channel tile (N) 0..3
    const int pg = (wave >> 2) * 8;   // first position tile (M)
    const int ln = lane & 15;
    const int hi = lane >> 4;         // lane half

    const float bias = b2s[ct * 16 + ln];
    v8f acc[8];
    #pragma unroll
    for (int i = 0; i < 8; ++i)
        acc[i] = (v8f){bias, bias, bias, bias, bias, bias, bias, bias};

    #pragma unroll
    for (int ks = 0; ks < 6; ++ks) {
        const int kb = ks * 32;
        // B (weights) fragment: col = ct*16+ln, element e -> K = kb + hi*16 + e (16 contiguous halfs)
        const _Float16* wp = &Ws[(ct * 16 + ln) * STR + kb + hi * 16];
        const v8h blo = *(const v8h*)wp;
        const v8h bhi = *(const v8h*)(wp + 8);
        const v16h bf = __builtin_shufflevector(blo, bhi,
            0,1,2,3,4,5,6,7,8,9,10,11,12,13,14,15);
        #pragma unroll
        for (int i = 0; i < 8; ++i) {
            // A (activations) fragment: row = pos, element e -> K = kb + hi*16 + e (16 contiguous halfs)
            const int row = (pg + i) * 16 + ln;
            const _Float16* ap = &Bm[row * STR + kb + hi * 16];
            const v8h alo = *(const v8h*)ap;
            const v8h ahi = *(const v8h*)(ap + 8);
            const v16h af = __builtin_shufflevector(alo, ahi,
                0,1,2,3,4,5,6,7,8,9,10,11,12,13,14,15);
            acc[i] = __builtin_amdgcn_wmma_f32_16x16x32_f16(
                false, af, false, bf, (short)0, acc[i], false, false);
        }
    }

    // Epilogue: GELU then per-channel (column) sum; C col = lane%16, rows = positions
    float s = 0.f;
    #pragma unroll
    for (int i = 0; i < 8; ++i) {
        #pragma unroll
        for (int r = 0; r < 8; ++r)
            s += gelu_erf(acc[i][r]);
    }
    s += __shfl_xor(s, 16, 32);       // merge row-halves (same channel in lane and lane^16)
    if (lane < 16) atomicAdd(&pooled[ct * 16 + lane], s);   // ds_add_f32
    __syncthreads();
    if (tid < HCH) pooled[tid] *= (1.0f / (float)PL);        // mean pool
    __syncthreads();

    // ---------------- Phase D: out = W3 @ pooled + b3  (768x64 matvec) ----------------
    float4 pr[16];
    {
        const float4* pv = (const float4*)pooled;
        #pragma unroll
        for (int q = 0; q < 16; ++q) pr[q] = pv[q];
    }
    #pragma unroll
    for (int j = 0; j < 3; ++j) {
        const int d = j * 256 + tid;
        const float4* wrow = (const float4*)(W3 + ((size_t)t * DM + d) * HCH);
        float a0 = b3[t * DM + d], a1 = 0.f, a2 = 0.f, a3 = 0.f;
        #pragma unroll
        for (int q = 0; q < 16; ++q) {
            const float4 w = wrow[q];
            a0 = fmaf(w.x, pr[q].x, a0);
            a1 = fmaf(w.y, pr[q].y, a1);
            a2 = fmaf(w.z, pr[q].z, a2);
            a3 = fmaf(w.w, pr[q].w, a3);
        }
        out[(size_t)m * DM + d] = (a0 + a1) + (a2 + a3);
    }
}

extern "C" void kernel_launch(void* const* d_in, const int* in_sizes, int n_in,
                              void* d_out, int out_size, void* d_ws, size_t ws_size,
                              hipStream_t stream) {
    const float* patches = (const float*)d_in[0];
    const int*   types   = (const int*)d_in[1];
    const float* W1      = (const float*)d_in[2];
    const float* b1      = (const float*)d_in[3];
    const float* W2      = (const float*)d_in[4];
    const float* b2      = (const float*)d_in[5];
    const float* W3      = (const float*)d_in[6];
    const float* b3      = (const float*)d_in[7];
    float* out = (float*)d_out;

    const int M = in_sizes[0] / PL;   // B*N = 8192 patches
    // LDS: Bm 256*200 + Ws 64*200 halfs = 128000B, xs 260f, pooled 64f, b2s 64f
    const size_t smem = (size_t)(PL + HCH) * STR * sizeof(_Float16)
                      + 260 * sizeof(float) + 2 * HCH * sizeof(float);

    hipLaunchKernelGGL(modality_stem_kernel, dim3(M), dim3(256), smem, stream,
                       patches, types, W1, b1, W2, b2, W3, b3, out);
}
